// GroupedQueryAttention_55087250539087
// MI455X (gfx1250) — matrix operand order repaired and checked by reading, not compiled, not run on hardware
//
#include <hip/hip_runtime.h>
#include <cstddef>
#include <cstdint>

// ---------------------------------------------------------------------------
// GQA block for MI455X (gfx1250, wave32, WMMA 16x16x32 f16 -> f32 accum)
// Pipeline: cvt weights -> rmsnorm -> QKV GEMM (+fused RoPE, V transpose)
//           -> flash attention -> out GEMM + residual
// GEMMs use double-buffered GLOBAL_LOAD_ASYNC_TO_LDS_B128 staging (ASYNCcnt),
// with the last K-step peeled so the steady-state loop is branch-free.
// ---------------------------------------------------------------------------

typedef __attribute__((ext_vector_type(16))) _Float16 v16h;
typedef __attribute__((ext_vector_type(8)))  _Float16 v8h;
typedef __attribute__((ext_vector_type(8)))  float    v8f;

#define DIM      2048
#define SEQ      2048
#define BATCH    2
#define NTOK     (BATCH * SEQ)     // 4096
#define CQKV     3072
#define NHEADS   32
#define NGROUPS  8
#define DHEAD    64

// Workspace layout (bytes)
#define WS_YH     ((size_t)0)                    // 4096*2048 f16
#define WS_WHIN   ((size_t)16777216)             // 3072*2048 f16
#define WS_WHOUT  ((size_t)29360128)             // 2048*2048 f16
#define WS_QH     ((size_t)37748736)             // 64*2048*64 f16 (rope'd)
#define WS_KH     ((size_t)54525952)             // 16*2048*64 f16 (rope'd)
#define WS_VTH    ((size_t)58720256)             // 16*64*2048 f16 (transposed)
#define WS_OH     ((size_t)62914560)             // 4096*2048 f16

__device__ __forceinline__ v8f wmma16(v16h a, v16h b, v8f c) {
  // D = A(16x32) * B(32x16) + C, f32 accumulate
  return __builtin_amdgcn_wmma_f32_16x16x32_f16(false, a, false, b, (short)0, c,
                                                false, false);
}

// A-matrix fragment: lane holds row (lane&15); 16 contiguous K halves at (lane>>4)*16.
__device__ __forceinline__ v16h load_af(const _Float16* p) {
  union { v16h v; v8h h[2]; } u;
  u.h[0] = *reinterpret_cast<const v8h*>(p);
  u.h[1] = *reinterpret_cast<const v8h*>(p + 8);
  return u.v;
}
// B-matrix fragment: lane holds col (lane&15); 16 contiguous K halves at (lane>>4)*16.
__device__ __forceinline__ v16h load_bf(const _Float16* p) {
  return *reinterpret_cast<const v16h*>(p);
}

// ---------------------------------------------------------------------------
// Shared GEMM core: block tile 128 (rows) x 256 (cols), K stepped by 32.
// A-tile 128x32 f16 (8KB) + B-tile 256x32 f16 (16KB) staged to LDS via
// async DMA, double buffered (48KB of the 320KB WGP LDS). Both A and B are
// row-major with K contiguous ("NT" GEMM). Per wave: 64x64 out = 16 accums.
// ---------------------------------------------------------------------------
#define BUF_HALFS 12288          // 24KB per buffer (A 4096 halfs + B 8192)
#define BUF_BYTES 24576

__device__ __forceinline__ void gemm_issue_tile(
    const _Float16* __restrict__ Ag, const _Float16* __restrict__ Bg,
    unsigned lbase, int tid, int buf, int k) {
  const unsigned lb = lbase + (unsigned)buf * BUF_BYTES;
  // A tile: 8KB = 512 x 16B chunks, 2 per thread. chunk c -> row c>>2,
  // byte offset (c&3)*16 within the 64B row.
#pragma unroll
  for (int i = 0; i < 2; i++) {
    const int c = tid + i * 256;
    const unsigned long long g = (unsigned long long)(uintptr_t)(
        (const char*)Ag + (size_t)(c >> 2) * (DIM * 2) + (size_t)k * 2 +
        (c & 3) * 16);
    const unsigned l = lb + (unsigned)c * 16u;
    asm volatile("global_load_async_to_lds_b128 %0, %1, off"
                 :: "v"(l), "v"(g) : "memory");
  }
  // B tile: 16KB = 1024 chunks, 4 per thread.
#pragma unroll
  for (int i = 0; i < 4; i++) {
    const int c = tid + i * 256;
    const unsigned long long g = (unsigned long long)(uintptr_t)(
        (const char*)Bg + (size_t)(c >> 2) * (DIM * 2) + (size_t)k * 2 +
        (c & 3) * 16);
    const unsigned l = lb + 8192u + (unsigned)c * 16u;
    asm volatile("global_load_async_to_lds_b128 %0, %1, off"
                 :: "v"(l), "v"(g) : "memory");
  }
}

__device__ __forceinline__ void gemm_compute_tile(
    const _Float16* lbuf, int cur, int wrow, int wcol, int lr, int hl,
    v8f acc[4][4]) {
  const _Float16* La = lbuf + (size_t)cur * BUF_HALFS;
  const _Float16* Lb = La + 4096;
  v16h af[4], bf[4];
#pragma unroll
  for (int a = 0; a < 4; a++)
    af[a] = load_af(La + (wrow * 64 + a * 16 + lr) * 32 + hl * 16);
#pragma unroll
  for (int c = 0; c < 4; c++)
    bf[c] = load_bf(Lb + (wcol * 64 + c * 16 + lr) * 32 + hl * 16);
#pragma unroll
  for (int a = 0; a < 4; a++)
#pragma unroll
    for (int c = 0; c < 4; c++) acc[a][c] = wmma16(af[a], bf[c], acc[a][c]);
}

__device__ __forceinline__ void gemm_core_async(
    const _Float16* __restrict__ Ag,   // block base: row (blockIdx.x*128), col 0
    const _Float16* __restrict__ Bg,   // block base: row (blockIdx.y*256), col 0
    _Float16* lbuf,                    // __shared__, 2*BUF_HALFS
    v8f acc[4][4]) {
  const int tid = threadIdx.x;
  const int lane = tid & 31, wave = tid >> 5;
  const int lr = lane & 15, hl = lane >> 4;
  const int wrow = wave & 1, wcol = wave >> 1;
  // Low 32 bits of a generic pointer to LDS == wave-relative LDS byte address.
  const unsigned lbase = (unsigned)(uintptr_t)lbuf;

  gemm_issue_tile(Ag, Bg, lbase, tid, 0, 0);
  int cur = 0;
  // Steady state: always prefetch the next tile; last iteration peeled below.
  for (int k = 0; k < DIM - 32; k += 32) {
    gemm_issue_tile(Ag, Bg, lbase, tid, cur ^ 1, k + 32);
    // Async loads complete in order: <=6 outstanding => previous tile resident.
    asm volatile("s_wait_asynccnt 6" ::: "memory");
    __syncthreads();
    gemm_compute_tile(lbuf, cur, wrow, wcol, lr, hl, acc);
    __syncthreads();   // all waves done reading before buffer is re-filled
    cur ^= 1;
  }
  asm volatile("s_wait_asynccnt 0" ::: "memory");
  __syncthreads();
  gemm_compute_tile(lbuf, cur, wrow, wcol, lr, hl, acc);
}

// ------------------------------- kernel 0 ----------------------------------
__global__ __launch_bounds__(256) void k_cvt_weights(
    const float* __restrict__ win, const float* __restrict__ wout,
    _Float16* __restrict__ whin, _Float16* __restrict__ whout) {
  const int NIN = CQKV * DIM;
  const int NOUT = DIM * NHEADS * DHEAD;
  int stride = gridDim.x * blockDim.x;
  for (int i = blockIdx.x * blockDim.x + threadIdx.x; i < NIN + NOUT; i += stride) {
    if (i < NIN) whin[i] = (_Float16)win[i];
    else         whout[i - NIN] = (_Float16)wout[i - NIN];
  }
}

// ------------------------------- kernel 1 ----------------------------------
__global__ __launch_bounds__(256) void k_rmsnorm(
    const float* __restrict__ x, const float* __restrict__ rmsw,
    _Float16* __restrict__ yh) {
  __shared__ float red[256];
  const int t = blockIdx.x;
  const float* xr = x + (size_t)t * DIM;
  float ss = 0.f;
  for (int c = threadIdx.x; c < DIM; c += 256) { float v = xr[c]; ss += v * v; }
  red[threadIdx.x] = ss;
  __syncthreads();
  for (int off = 128; off > 0; off >>= 1) {
    if ((int)threadIdx.x < off) red[threadIdx.x] += red[threadIdx.x + off];
    __syncthreads();
  }
  const float rr = rsqrtf(red[0] * (1.0f / DIM) + 1e-6f);
  for (int c = threadIdx.x; c < DIM; c += 256)
    yh[(size_t)t * DIM + c] = (_Float16)(xr[c] * rr * rmsw[c]);
}

// ------------------------------- kernel 2 ----------------------------------
// qkv[t][c] = sum_m y[t][m] * w_in[c][m]; fused RoPE epilogue; V transposed.
__global__ __launch_bounds__(256) void k_qkv_gemm_rope(
    const _Float16* __restrict__ yh, const _Float16* __restrict__ wh,
    _Float16* __restrict__ Qh, _Float16* __restrict__ Kh,
    _Float16* __restrict__ Vth) {
  __shared__ __align__(64) _Float16 lbuf[2 * BUF_HALFS];
  const int lane = threadIdx.x & 31, wave = threadIdx.x >> 5;
  const int lr = lane & 15, hl = lane >> 4;
  const int rowBase = blockIdx.x * 128 + (wave & 1) * 64;
  const int colBase = blockIdx.y * 256 + (wave >> 1) * 64;

  const v8f zero = {0.f, 0.f, 0.f, 0.f, 0.f, 0.f, 0.f, 0.f};
  v8f acc[4][4];
#pragma unroll
  for (int a = 0; a < 4; a++)
#pragma unroll
    for (int c = 0; c < 4; c++) acc[a][c] = zero;

  gemm_core_async(yh + (size_t)(blockIdx.x * 128) * DIM,
                  wh + (size_t)(blockIdx.y * 256) * DIM, lbuf, acc);

  // Epilogue. Accum layout: lane, reg r -> row M = r + 8*hl, col N = lr.
  if (colBase < 2048) {                      // --- Q path (rope) ---
    const int head = colBase >> 6;
#pragma unroll
    for (int c = 0; c < 2; c++) {
      const int d = c * 16 + lr;             // d in [0,32)
      const float invf = __powf(10000.0f, -(float)d * (1.0f / 32.0f));
#pragma unroll
      for (int a = 0; a < 4; a++)
#pragma unroll
        for (int r = 0; r < 8; r++) {
          const int t = rowBase + a * 16 + r + hl * 8;
          const int bb = t >> 11, spos = t & (SEQ - 1);
          float si, co;
          __sincosf((float)spos * invf, &si, &co);
          const float v1 = acc[a][c][r], v2 = acc[a][c + 2][r];
          _Float16* q = Qh + ((size_t)(bb * NHEADS + head) * SEQ + spos) * DHEAD;
          q[d]      = (_Float16)(v1 * co - v2 * si);
          q[d + 32] = (_Float16)(v2 * co + v1 * si);
        }
    }
  } else if (colBase < 2560) {               // --- K path (rope) ---
    const int g = (colBase - 2048) >> 6;
#pragma unroll
    for (int c = 0; c < 2; c++) {
      const int d = c * 16 + lr;
      const float invf = __powf(10000.0f, -(float)d * (1.0f / 32.0f));
#pragma unroll
      for (int a = 0; a < 4; a++)
#pragma unroll
        for (int r = 0; r < 8; r++) {
          const int t = rowBase + a * 16 + r + hl * 8;
          const int bb = t >> 11, spos = t & (SEQ - 1);
          float si, co;
          __sincosf((float)spos * invf, &si, &co);
          const float v1 = acc[a][c][r], v2 = acc[a][c + 2][r];
          _Float16* kk = Kh + ((size_t)(bb * NGROUPS + g) * SEQ + spos) * DHEAD;
          kk[d]      = (_Float16)(v1 * co - v2 * si);
          kk[d + 32] = (_Float16)(v2 * co + v1 * si);
        }
    }
  } else {                                   // --- V path (transpose store) ---
    const int g = (colBase - 2560) >> 6;
#pragma unroll
    for (int c = 0; c < 4; c++) {
      const int d = c * 16 + lr;
#pragma unroll
      for (int a = 0; a < 4; a++)
#pragma unroll
        for (int r = 0; r < 8; r++) {
          const int t = rowBase + a * 16 + r + hl * 8;
          const int bb = t >> 11, spos = t & (SEQ - 1);
          Vth[((size_t)(bb * NGROUPS + g) * DHEAD + d) * SEQ + spos] =
              (_Float16)acc[a][c][r];
        }
    }
  }
}

// ------------------------------- kernel 3 ----------------------------------
// Flash attention, causal. One wave = 16 query rows; 32-key tiles.
__global__ __launch_bounds__(256) void k_attention(
    const _Float16* __restrict__ Qh, const _Float16* __restrict__ Kh,
    const _Float16* __restrict__ Vth, _Float16* __restrict__ oh) {
  __shared__ __align__(32) _Float16 plds[8][16 * 32];  // per-wave P tile
  const int lane = threadIdx.x & 31, wave = threadIdx.x >> 5;
  const int lr = lane & 15, hl = lane >> 4;
  const int bh = blockIdx.y;                  // 0..63
  const int b = bh >> 5, head = bh & 31;
  const int kvi = b * NGROUPS + (head >> 2);
  const int qbase = blockIdx.x * 128 + wave * 16;

  const _Float16* Qrow = Qh + ((size_t)bh * SEQ + qbase + lr) * DHEAD;
  const v16h qf0 = load_af(Qrow + hl * 16);        // d slice [0,32)
  const v16h qf1 = load_af(Qrow + 32 + hl * 16);   // d slice [32,64)

  const _Float16* Kbase = Kh + (size_t)kvi * SEQ * DHEAD;
  const _Float16* Vbase = Vth + (size_t)kvi * DHEAD * SEQ;
  _Float16* pl = &plds[wave][0];

  const v8f zero = {0.f, 0.f, 0.f, 0.f, 0.f, 0.f, 0.f, 0.f};
  float m[8], l[8];
  v8f O[4];
#pragma unroll
  for (int r = 0; r < 8; r++) { m[r] = -1e30f; l[r] = 0.f; }
#pragma unroll
  for (int dt = 0; dt < 4; dt++) O[dt] = zero;

  const int jend = qbase + 16;  // keys <= last query row of this wave
  for (int j = 0; j < jend; j += 32) {
    // S = Q K^T (two 16-key subtiles, two 32-wide d slices each)
    const _Float16* K0 = Kbase + (size_t)(j + lr) * DHEAD + hl * 16;
    const _Float16* K1 = K0 + 16 * DHEAD;
    v8f S0 = zero, S1 = zero;
    S0 = wmma16(qf0, load_bf(K0), S0);
    S0 = wmma16(qf1, load_bf(K0 + 32), S0);
    S1 = wmma16(qf0, load_bf(K1), S1);
    S1 = wmma16(qf1, load_bf(K1 + 32), S1);

    // online softmax update per row (reg r owns row r + 8*hl)
#pragma unroll
    for (int r = 0; r < 8; r++) {
      float s0 = S0[r] * 0.125f;  // 1/sqrt(64)
      float s1 = S1[r] * 0.125f;
      const int rowA = qbase + r + hl * 8;
      const int c0 = j + lr;
      if (c0 > rowA) s0 = -1e30f;
      if (c0 + 16 > rowA) s1 = -1e30f;
      float mx = fmaxf(s0, s1);
      mx = fmaxf(mx, __shfl_xor(mx, 1));
      mx = fmaxf(mx, __shfl_xor(mx, 2));
      mx = fmaxf(mx, __shfl_xor(mx, 4));
      mx = fmaxf(mx, __shfl_xor(mx, 8));
      const float mn = fmaxf(m[r], mx);
      const float p0 = __expf(s0 - mn);
      const float p1 = __expf(s1 - mn);
      float rs = p0 + p1;
      rs += __shfl_xor(rs, 1);
      rs += __shfl_xor(rs, 2);
      rs += __shfl_xor(rs, 4);
      rs += __shfl_xor(rs, 8);
      const float al = __expf(m[r] - mn);
      l[r] = l[r] * al + rs;
      m[r] = mn;
      O[0][r] *= al; O[1][r] *= al; O[2][r] *= al; O[3][r] *= al;
      const int prow = r + hl * 8;
      pl[prow * 32 + lr] = (_Float16)p0;
      pl[prow * 32 + 16 + lr] = (_Float16)p1;
    }

    // P (16x32) as A fragment via LDS transpose; O += P * V
    const v16h pf = load_af(pl + lr * 32 + hl * 16);
#pragma unroll
    for (int dt = 0; dt < 4; dt++) {
      const _Float16* Vp = Vbase + (size_t)(dt * 16 + lr) * SEQ + j + hl * 16;
      O[dt] = wmma16(pf, load_bf(Vp), O[dt]);
    }
  }

  // normalize and store o[t][head*64 + d]
#pragma unroll
  for (int r = 0; r < 8; r++) {
    const float invl = 1.0f / l[r];
    const int t = b * SEQ + qbase + r + hl * 8;
#pragma unroll
    for (int dt = 0; dt < 4; dt++)
      oh[(size_t)t * (NHEADS * DHEAD) + head * DHEAD + dt * 16 + lr] =
          (_Float16)(O[dt][r] * invl);
  }
}

// ------------------------------- kernel 4 ----------------------------------
// out[t][m] = x[t][m] + sum_c o[t][c] * w_out[m][c]
__global__ __launch_bounds__(256) void k_out_gemm(
    const _Float16* __restrict__ oh, const _Float16* __restrict__ wh,
    const float* __restrict__ x, float* __restrict__ out) {
  __shared__ __align__(64) _Float16 lbuf[2 * BUF_HALFS];
  const int lane = threadIdx.x & 31, wave = threadIdx.x >> 5;
  const int lr = lane & 15, hl = lane >> 4;
  const int rowBase = blockIdx.x * 128 + (wave & 1) * 64;
  const int colBase = blockIdx.y * 256 + (wave >> 1) * 64;

  const v8f zero = {0.f, 0.f, 0.f, 0.f, 0.f, 0.f, 0.f, 0.f};
  v8f acc[4][4];
#pragma unroll
  for (int a = 0; a < 4; a++)
#pragma unroll
    for (int c = 0; c < 4; c++) acc[a][c] = zero;

  gemm_core_async(oh + (size_t)(blockIdx.x * 128) * DIM,
                  wh + (size_t)(blockIdx.y * 256) * DIM, lbuf, acc);

#pragma unroll
  for (int a = 0; a < 4; a++)
#pragma unroll
    for (int c = 0; c < 4; c++)
#pragma unroll
      for (int r = 0; r < 8; r++) {
        const int t = rowBase + a * 16 + r + hl * 8;
        const int mm = colBase + c * 16 + lr;
        out[(size_t)t * DIM + mm] = x[(size_t)t * DIM + mm] + acc[a][c][r];
      }
}

// ------------------------------- launch ------------------------------------
extern "C" void kernel_launch(void* const* d_in, const int* in_sizes, int n_in,
                              void* d_out, int out_size, void* d_ws, size_t ws_size,
                              hipStream_t stream) {
  const float* x     = (const float*)d_in[0];
  const float* w_in  = (const float*)d_in[1];
  const float* w_out = (const float*)d_in[2];
  const float* rms_w = (const float*)d_in[3];
  float* out = (float*)d_out;

  char* ws = (char*)d_ws;
  _Float16* yh    = (_Float16*)(ws + WS_YH);
  _Float16* whin  = (_Float16*)(ws + WS_WHIN);
  _Float16* whout = (_Float16*)(ws + WS_WHOUT);
  _Float16* Qh    = (_Float16*)(ws + WS_QH);
  _Float16* Kh    = (_Float16*)(ws + WS_KH);
  _Float16* Vth   = (_Float16*)(ws + WS_VTH);
  _Float16* oh    = (_Float16*)(ws + WS_OH);

  k_cvt_weights<<<2048, 256, 0, stream>>>(w_in, w_out, whin, whout);
  k_rmsnorm<<<NTOK, 256, 0, stream>>>(x, rms_w, yh);
  k_qkv_gemm_rope<<<dim3(NTOK / 128, CQKV / 256), 256, 0, stream>>>(
      yh, whin, Qh, Kh, Vth);
  k_attention<<<dim3(SEQ / 128, BATCH * NHEADS), 256, 0, stream>>>(
      Qh, Kh, Vth, oh);
  k_out_gemm<<<dim3(NTOK / 128, DIM / 256), 256, 0, stream>>>(
      oh, whout, x, out);
}
